// TopDownCore_VFVA_88003879895244
// MI455X (gfx1250) — compile-verified, matrix operand order checked by reading, not run on hardware
//
#include <hip/hip_runtime.h>
#include <hip/hip_bf16.h>

// ---------------- problem constants (from reference) ----------------
constexpr int Bc   = 512;
constexpr int Ec   = 1024;
constexpr int Rc   = 1024;
constexpr int Ac   = 512;
constexpr int NVc  = 196;
constexpr int SELc = 36;

typedef __attribute__((ext_vector_type(16))) __bf16 v16bf;
typedef __attribute__((ext_vector_type(8)))  float  v8f;
typedef __attribute__((ext_vector_type(4)))  int    v4i;

#if defined(__has_builtin)
#  if __has_builtin(__builtin_amdgcn_global_load_async_to_lds_b128)
#    define USE_ASYNC_LDS 1
#  endif
#  if __has_builtin(__builtin_amdgcn_s_wait_asynccnt)
#    define HAVE_WAIT_ASYNC_BUILTIN 1
#  endif
#endif

// ---------------- WMMA GEMM:  C[M,N] = A[M,K](bf16) * W[N,K]^T (+b0+b1) (+=C) ----
// Requirements used by this pipeline: K % 32 == 0, M % 64 == 0 (M is 512 or 18432).
#define TM 64
#define TN 64
#define TK 32

__global__ __launch_bounds__(256)
void gemm_xwT(const __bf16* __restrict__ A, const float* __restrict__ W,
              const float* __restrict__ bias0, const float* __restrict__ bias1,
              float* __restrict__ C, int M, int N, int K, int addC)
{
    __shared__ __align__(16) __bf16 As[TM][TK];
    __shared__ __align__(16) __bf16 Bs[TN][TK];

    const int tiles_n = (N + TN - 1) / TN;
    const int bm = blockIdx.x / tiles_n;
    const int bn = blockIdx.x % tiles_n;
    const int m0 = bm * TM, n0 = bn * TN;

    const int tid  = threadIdx.x;
    const int lane = tid & 31;
    const int wave = tid >> 5;
    const int wm   = (wave & 3) * 16;   // wave row offset in tile
    const int wn   = (wave >> 2) * 32;  // wave col offset in tile (2 N-tiles)
    const int half = lane >> 4;
    const int r    = lane & 15;

    // per-thread staging coordinates: 16 bytes each, 256 threads cover a 64x32 tile
    const int smm = tid >> 2;           // row in tile (A: m, B: n)
    const int skk = (tid & 3) * 8;      // 8-element (16B) column chunk

    v8f acc0 = {}; v8f acc1 = {};

    for (int k0 = 0; k0 < K; k0 += TK) {
        // ---- A tile: bf16, no conversion needed -> async copy global->LDS ----
        {
            const __bf16* gsrc = A + (size_t)(m0 + smm) * K + (k0 + skk);
            __bf16* ldst = &As[smm][skk];
#ifdef USE_ASYNC_LDS
            __builtin_amdgcn_global_load_async_to_lds_b128(
                (__attribute__((address_space(1))) v4i*)gsrc,
                (__attribute__((address_space(3))) v4i*)ldst, 0, 0);
#else
            *(uint4*)ldst = *(const uint4*)gsrc;
#endif
        }
        // ---- W tile: f32 -> bf16 conversion through VGPRs, 16B LDS store ----
        {
            int gn = n0 + smm;
            float4 w0 = make_float4(0.f, 0.f, 0.f, 0.f);
            float4 w1 = w0;
            if (gn < N) {
                const float* gw = W + (size_t)gn * K + (k0 + skk);
                w0 = *(const float4*)gw;
                w1 = *(const float4*)(gw + 4);
            }
            __bf16* ldst = &Bs[smm][skk];
            ldst[0] = (__bf16)w0.x; ldst[1] = (__bf16)w0.y;
            ldst[2] = (__bf16)w0.z; ldst[3] = (__bf16)w0.w;
            ldst[4] = (__bf16)w1.x; ldst[5] = (__bf16)w1.y;
            ldst[6] = (__bf16)w1.z; ldst[7] = (__bf16)w1.w;
        }
        // prefetch next W K-slab into L2 (global_prefetch_b8)
        if (k0 + TK < K) {
            int pn = n0 + (tid & 63);
            if (pn < N) __builtin_prefetch(&W[(size_t)pn * K + k0 + TK]);
        }
#ifdef USE_ASYNC_LDS
#  ifdef HAVE_WAIT_ASYNC_BUILTIN
        __builtin_amdgcn_s_wait_asynccnt(0);
#  else
        asm volatile("s_wait_asynccnt 0x0" ::: "memory");
#  endif
#endif
        __syncthreads();

        // build fragments per ISA 16-bit 16x32 layout
        v16bf afrag, b0frag, b1frag;
#pragma unroll
        for (int e = 0; e < 16; ++e) {
            int vg = e >> 1, j = e & 1;
            int kk = (vg < 4) ? (half * 8 + vg * 2 + j)
                              : (16 + half * 8 + (vg - 4) * 2 + j);
            afrag[e]  = As[wm + r][kk];
            b0frag[e] = Bs[wn + r][kk];
            b1frag[e] = Bs[wn + 16 + r][kk];
        }
        acc0 = __builtin_amdgcn_wmma_f32_16x16x32_bf16(false, afrag, false, b0frag,
                                                       (short)0, acc0, false, false);
        acc1 = __builtin_amdgcn_wmma_f32_16x16x32_bf16(false, afrag, false, b1frag,
                                                       (short)0, acc1, false, false);
        __syncthreads();
    }

    // epilogue: C layout = lane&15 -> N, vgpr + 8*half -> M
    const int cn = r;
    const int cm = half * 8;
#pragma unroll
    for (int v = 0; v < 8; ++v) {
        int gm = m0 + wm + cm + v;
        if (gm >= M) continue;
        int gn0 = n0 + wn + cn;
        int gn1 = gn0 + 16;
        if (gn0 < N) {
            float val = acc0[v];
            if (bias0) val += bias0[gn0];
            if (bias1) val += bias1[gn0];
            if (addC)  val += C[(size_t)gm * N + gn0];
            C[(size_t)gm * N + gn0] = val;
        }
        if (gn1 < N) {
            float val = acc1[v];
            if (bias0) val += bias0[gn1];
            if (bias1) val += bias1[gn1];
            if (addC)  val += C[(size_t)gm * N + gn1];
            C[(size_t)gm * N + gn1] = val;
        }
    }
}

// ---------------- concat up to 3 f32 sources -> bf16 [M, k0+k1+k2] ----------------
__global__ void pack3_bf16(const float* __restrict__ s0, int k0,
                           const float* __restrict__ s1, int k1,
                           const float* __restrict__ s2, int k2,
                           __bf16* __restrict__ dst, int M)
{
    int Ktot = k0 + k1 + k2;
    size_t idx = (size_t)blockIdx.x * blockDim.x + threadIdx.x;
    size_t total = (size_t)M * Ktot;
    if (idx >= total) return;
    int m = (int)(idx / Ktot), k = (int)(idx % Ktot);
    float v;
    if (k < k0)            v = s0[(size_t)m * k0 + k];
    else if (k < k0 + k1)  v = s1[(size_t)m * k1 + (k - k0)];
    else                   v = s2[(size_t)m * k2 + (k - k0 - k1)];
    dst[idx] = (__bf16)v;
}

// ---------------- LSTM cell elementwise (gate order i,f,g,o) ----------------
__global__ void lstm_cell_k(const float* __restrict__ G, const float* __restrict__ c_in,
                            float* __restrict__ h_out, float* __restrict__ c_out,
                            __bf16* __restrict__ h_bf, float* __restrict__ h_copy)
{
    int idx = blockIdx.x * blockDim.x + threadIdx.x;
    if (idx >= Bc * Rc) return;
    int b = idx / Rc, rr = idx % Rc;
    const float* g = G + (size_t)b * 4 * Rc;
    float gi = g[rr], gf = g[Rc + rr], gg = g[2 * Rc + rr], go = g[3 * Rc + rr];
    float si = 1.0f / (1.0f + __expf(-gi));
    float sf = 1.0f / (1.0f + __expf(-gf));
    float so = 1.0f / (1.0f + __expf(-go));
    float c2 = sf * c_in[idx] + si * tanhf(gg);
    float h2 = so * tanhf(c2);
    c_out[idx] = c2;
    h_out[idx] = h2;
    if (h_copy) h_copy[idx] = h2;
    if (h_bf)   h_bf[idx]   = (__bf16)h2;
}

// ---------- scores[b,s] = sum_a tanh(p[b,s,a] + ah[b,a]) * wal[a] + bal -----------
__global__ void att_scores_k(const float* __restrict__ p, const float* __restrict__ ah,
                             const float* __restrict__ wal, const float* __restrict__ bal,
                             float* __restrict__ scores, int S)
{
    int wid  = (int)(((size_t)blockIdx.x * blockDim.x + threadIdx.x) >> 5);
    int lane = threadIdx.x & 31;
    if (wid >= Bc * S) return;
    int b = wid / S;
    const float* pr = p  + (size_t)wid * Ac;
    const float* ar = ah + (size_t)b   * Ac;
    float sum = 0.0f;
    for (int a = lane; a < Ac; a += 32)
        sum += tanhf(pr[a] + ar[a]) * wal[a];
#pragma unroll
    for (int off = 16; off; off >>= 1) sum += __shfl_xor(sum, off, 32);
    if (lane == 0) scores[wid] = sum + bal[0];
}

// ---------------- row softmax over S ----------------
__global__ void softmax_rows_k(const float* __restrict__ x, float* __restrict__ y, int S)
{
    int b = blockIdx.x;
    __shared__ float red[256];
    const float* xr = x + (size_t)b * S;
    float m = -1e30f;
    for (int s = threadIdx.x; s < S; s += 256) m = fmaxf(m, xr[s]);
    red[threadIdx.x] = m; __syncthreads();
    for (int st = 128; st; st >>= 1) {
        if (threadIdx.x < st) red[threadIdx.x] = fmaxf(red[threadIdx.x], red[threadIdx.x + st]);
        __syncthreads();
    }
    m = red[0]; __syncthreads();
    float sum = 0.0f;
    for (int s = threadIdx.x; s < S; s += 256) sum += __expf(xr[s] - m);
    red[threadIdx.x] = sum; __syncthreads();
    for (int st = 128; st; st >>= 1) {
        if (threadIdx.x < st) red[threadIdx.x] += red[threadIdx.x + st];
        __syncthreads();
    }
    float inv = 1.0f / red[0];
    for (int s = threadIdx.x; s < S; s += 256)
        y[(size_t)b * S + s] = __expf(xr[s] - m) * inv;
}

// ------------- out[b,:] = sum_s w[b,s] * feats[b,s,:]  (f32 feats) ---------------
__global__ void wsum_f32_k(const float* __restrict__ w, const float* __restrict__ feats,
                           float* __restrict__ out, int S)
{
    int b = blockIdx.x;
    for (int d = threadIdx.x; d < Rc; d += 256) {
        float acc = 0.0f;
        const float* f = feats + (size_t)b * S * Rc + d;
        for (int s = 0; s < S; ++s) acc += w[(size_t)b * S + s] * f[(size_t)s * Rc];
        out[(size_t)b * Rc + d] = acc;
    }
}

// ------------- same, bf16 feats ---------------
__global__ void wsum_bf16_k(const float* __restrict__ w, const __bf16* __restrict__ feats,
                            float* __restrict__ out, int S)
{
    int b = blockIdx.x;
    for (int d = threadIdx.x; d < Rc; d += 256) {
        float acc = 0.0f;
        const __bf16* f = feats + (size_t)b * S * Rc + d;
        for (int s = 0; s < S; ++s) acc += w[(size_t)b * S + s] * (float)f[(size_t)s * Rc];
        out[(size_t)b * Rc + d] = acc;
    }
}

// ------------- top-36 of 196 logits (descending, lower index on tie) -------------
__global__ void topk36_k(const float* __restrict__ logits, int* __restrict__ ix)
{
    int b = blockIdx.x;
    __shared__ float vals[256];
    __shared__ float rv[256];
    __shared__ int   ri[256];
    int t = threadIdx.x;
    vals[t] = (t < NVc) ? logits[(size_t)b * NVc + t] : -1e30f;
    __syncthreads();
    for (int k = 0; k < SELc; ++k) {
        rv[t] = vals[t]; ri[t] = t;
        __syncthreads();
        for (int st = 128; st; st >>= 1) {
            if (t < st) {
                if (rv[t + st] > rv[t] || (rv[t + st] == rv[t] && ri[t + st] < ri[t])) {
                    rv[t] = rv[t + st]; ri[t] = ri[t + st];
                }
            }
            __syncthreads();
        }
        if (t == 0) { ix[b * SELc + k] = ri[0]; vals[ri[0]] = -1e30f; }
        __syncthreads();
    }
}

// ------------- gather arr1[b,s,:] = bf16(att_feats[b, ix[b,s], :]) -------------
__global__ void gather_rows_bf16_k(const float* __restrict__ feats, const int* __restrict__ ix,
                                   __bf16* __restrict__ out)
{
    int bs  = blockIdx.x;              // b*SEL + s
    int b   = bs / SELc;
    int row = ix[bs];
    const float* src = feats + ((size_t)b * NVc + row) * Rc;
    __bf16*      dst = out + (size_t)bs * Rc;
    for (int d = threadIdx.x; d < Rc; d += 256) dst[d] = (__bf16)src[d];
}

// ---------------- host-side launch helpers ----------------
static inline void launch_gemm(const __bf16* A, const float* W, const float* b0,
                               const float* b1, float* C, int M, int N, int K,
                               int addC, hipStream_t s)
{
    int tm = (M + TM - 1) / TM, tn = (N + TN - 1) / TN;
    gemm_xwT<<<dim3(tm * tn), dim3(256), 0, s>>>(A, W, b0, b1, C, M, N, K, addC);
}

extern "C" void kernel_launch(void* const* d_in, const int* in_sizes, int n_in,
                              void* d_out, int out_size, void* d_ws, size_t ws_size,
                              hipStream_t stream)
{
    const float* xt        = (const float*)d_in[0];
    const float* fc_feats  = (const float*)d_in[1];
    const float* att_feats = (const float*)d_in[2];
    const float* p_att     = (const float*)d_in[3];
    const float* sh        = (const float*)d_in[4];   // [2,B,R]
    const float* sc        = (const float*)d_in[5];   // [2,B,R]
    const float* W_ih_att  = (const float*)d_in[6];
    const float* b_ih_att  = (const float*)d_in[7];
    const float* W_hh_att  = (const float*)d_in[8];
    const float* b_hh_att  = (const float*)d_in[9];
    const float* W_ih_lang = (const float*)d_in[10];
    const float* b_ih_lang = (const float*)d_in[11];
    const float* W_hh_lang = (const float*)d_in[12];
    const float* b_hh_lang = (const float*)d_in[13];
    const float* Wh2att1   = (const float*)d_in[14];
    const float* bh2att1   = (const float*)d_in[15];
    const float* Walpha1   = (const float*)d_in[16];
    const float* balpha1   = (const float*)d_in[17];
    const float* Wh2att2   = (const float*)d_in[18];
    const float* bh2att2   = (const float*)d_in[19];
    const float* Walpha2   = (const float*)d_in[20];
    const float* balpha2   = (const float*)d_in[21];
    const float* W_vf      = (const float*)d_in[22];
    const float* b_vf      = (const float*)d_in[23];
    const float* W_pvf     = (const float*)d_in[24];
    const float* b_pvf     = (const float*)d_in[25];

    float* out = (float*)d_out;
    const size_t BR = (size_t)Bc * Rc;
    // out layout: [0,BR)=h_lang(output) [BR,2BR)=h_att [2BR,3BR)=h_lang [3BR,4BR)=c_att [4BR,5BR)=c_lang
    float* h_att_f  = out + BR;
    float* h_lang_f = out + 2 * BR;
    float* c_att_f  = out + 3 * BR;
    float* c_lang_f = out + 4 * BR;

    // workspace carve-out
    char* wp = (char*)d_ws;
    auto alloc = [&](size_t bytes) -> void* {
        void* p = (void*)wp;
        wp += (bytes + 255) & ~(size_t)255;
        return p;
    };
    __bf16* X1bf   = (__bf16*)alloc((size_t)Bc * 3072 * 2);
    __bf16* H0bf   = (__bf16*)alloc((size_t)Bc * Rc * 2);
    float*  G      = (float*) alloc((size_t)Bc * 4096 * 4);   // reused att/lang
    __bf16* hattbf = (__bf16*)alloc((size_t)Bc * Rc * 2);
    float*  atth1  = (float*) alloc((size_t)Bc * Ac * 4);
    float*  atth2  = (float*) alloc((size_t)Bc * Ac * 4);
    float*  sc1    = (float*) alloc((size_t)Bc * NVc * 4);
    float*  wsm1   = (float*) alloc((size_t)Bc * NVc * 4);
    float*  att_f  = (float*) alloc((size_t)Bc * Rc * 4);
    __bf16* concbf = (__bf16*)alloc((size_t)Bc * 2048 * 2);
    float*  vfha   = (float*) alloc((size_t)Bc * NVc * 4);
    int*    vfix   = (int*)   alloc((size_t)Bc * SELc * 4);
    __bf16* arr1bf = (__bf16*)alloc((size_t)Bc * SELc * Rc * 2);
    float*  pvf    = (float*) alloc((size_t)Bc * SELc * Ac * 4);
    float*  sc2    = (float*) alloc((size_t)Bc * SELc * 4);
    float*  wsm2   = (float*) alloc((size_t)Bc * SELc * 4);
    float*  vfatt  = (float*) alloc((size_t)Bc * Rc * 4);
    __bf16* X2bf   = (__bf16*)alloc((size_t)Bc * 3072 * 2);
    __bf16* H1bf   = (__bf16*)alloc((size_t)Bc * Rc * 2);

    auto grid1d = [](size_t total) { return dim3((unsigned)((total + 255) / 256)); };

    // ---- att LSTM ----
    pack3_bf16<<<grid1d((size_t)Bc * 3072), 256, 0, stream>>>(
        sh + BR, Rc, fc_feats, Rc, xt, Ec, X1bf, Bc);             // [prev_h, fc, xt]
    pack3_bf16<<<grid1d((size_t)Bc * Rc), 256, 0, stream>>>(
        sh, Rc, sh, 0, sh, 0, H0bf, Bc);                          // state_h[0]
    launch_gemm(X1bf, W_ih_att, b_ih_att, b_hh_att, G, Bc, 4096, 3072, 0, stream);
    launch_gemm(H0bf, W_hh_att, nullptr, nullptr,  G, Bc, 4096, 1024, 1, stream);
    lstm_cell_k<<<grid1d(BR), 256, 0, stream>>>(G, sc, h_att_f, c_att_f, hattbf, nullptr);

    // ---- attention 1 ----
    launch_gemm(hattbf, Wh2att1, bh2att1, nullptr, atth1, Bc, Ac, Rc, 0, stream);
    launch_gemm(hattbf, Wh2att2, bh2att2, nullptr, atth2, Bc, Ac, Rc, 0, stream);
    att_scores_k<<<grid1d((size_t)Bc * NVc * 32), 256, 0, stream>>>(
        p_att, atth1, Walpha1, balpha1, sc1, NVc);
    softmax_rows_k<<<Bc, 256, 0, stream>>>(sc1, wsm1, NVc);
    wsum_f32_k<<<Bc, 256, 0, stream>>>(wsm1, att_feats, att_f, NVc);

    // ---- vf selection ----
    pack3_bf16<<<grid1d((size_t)Bc * 2048), 256, 0, stream>>>(
        att_f, Rc, h_att_f, Rc, att_f, 0, concbf, Bc);            // [att, h_att]
    launch_gemm(concbf, W_vf, b_vf, nullptr, vfha, Bc, NVc, 2048, 0, stream);
    topk36_k<<<Bc, 256, 0, stream>>>(vfha, vfix);                 // top-k on logits == on softmax
    gather_rows_bf16_k<<<Bc * SELc, 256, 0, stream>>>(att_feats, vfix, arr1bf);
    launch_gemm(arr1bf, W_pvf, b_pvf, nullptr, pvf, Bc * SELc, Ac, Rc, 0, stream);

    // ---- vf attention ----
    att_scores_k<<<grid1d((size_t)Bc * SELc * 32), 256, 0, stream>>>(
        pvf, atth2, Walpha2, balpha2, sc2, SELc);
    softmax_rows_k<<<Bc, 256, 0, stream>>>(sc2, wsm2, SELc);
    wsum_bf16_k<<<Bc, 256, 0, stream>>>(wsm2, arr1bf, vfatt, SELc);

    // ---- lang LSTM ----
    pack3_bf16<<<grid1d((size_t)Bc * 3072), 256, 0, stream>>>(
        vfatt, Rc, h_att_f, Rc, att_f, Rc, X2bf, Bc);             // [vf_att, h_att, att]
    pack3_bf16<<<grid1d((size_t)Bc * Rc), 256, 0, stream>>>(
        sh + BR, Rc, sh, 0, sh, 0, H1bf, Bc);                     // state_h[1]
    launch_gemm(X2bf, W_ih_lang, b_ih_lang, b_hh_lang, G, Bc, 4096, 3072, 0, stream);
    launch_gemm(H1bf, W_hh_lang, nullptr, nullptr,   G, Bc, 4096, 1024, 1, stream);
    lstm_cell_k<<<grid1d(BR), 256, 0, stream>>>(G, sc + BR, h_lang_f, c_lang_f,
                                                nullptr, out /* output copy */);
}
